// MultiLayerAttention_37709812858884
// MI455X (gfx1250) — matrix operand order repaired and checked by reading, not compiled, not run on hardware
//
#include <hip/hip_runtime.h>
#include <stdint.h>

#define NB   2
#define SLEN 2048
#define NH   16
#define HD   128
#define EMB  2048

typedef __bf16 bf16_t;
typedef __attribute__((ext_vector_type(16))) bf16_t    v16bf;
typedef __attribute__((ext_vector_type(8)))  float     v8f;
typedef __attribute__((ext_vector_type(4)))  uint32_t  u32x4;
typedef __attribute__((ext_vector_type(8)))  uint32_t  u32x8;

union Frag16 {
    v16bf v;
    uint4 q[2];
    unsigned short u[16];
};

__device__ __forceinline__ unsigned short f2bf(float f) {
    uint32_t b = __float_as_uint(f);
    b += 0x7FFFu + ((b >> 16) & 1u);   // round-to-nearest-even
    return (unsigned short)(b >> 16);
}

__device__ __forceinline__ uint32_t lds_byte_addr(const void* p) {
    return (uint32_t)(uintptr_t)p;     // low 32 bits of flat shared addr = LDS offset
}

// ---------------------------------------------------------------------------
// Tensor Data Mover descriptor (2D tile, bf16).  group1 + lds_addr are
// loop-invariant; issue() patches only the global-address dwords, so the
// per-iteration cost is ~2 SALU ops + the tensor_load_to_lds itself.
// ---------------------------------------------------------------------------
struct Tdm2D {
    u32x4 g0;
    u32x8 g1;
    __device__ __forceinline__ void init(uint32_t lds_addr, uint32_t t0, uint32_t t1,
                                         uint32_t stride) {
        g0[0] = 1u;                                              // count=1
        g0[1] = lds_addr;                                        // lds_addr[63:32]
        g0[2] = 0u;
        g0[3] = 0u;
        g1[0] = 1u << 16;                                        // data_size=2B
        g1[1] = (t0 & 0xFFFFu) << 16;                            // tensor_dim0 lo16
        g1[2] = ((t0 >> 16) & 0xFFFFu) | ((t1 & 0xFFFFu) << 16); // dim0 hi | dim1 lo
        g1[3] = ((t1 >> 16) & 0xFFFFu) | ((t0 & 0xFFFFu) << 16); // dim1 hi | tile_dim0
        g1[4] = (t1 & 0xFFFFu);                                  // tile_dim1 (tile_dim2=0)
        g1[5] = stride;                                          // dim0_stride lo32
        g1[6] = 0u;
        g1[7] = 0u;
    }
    __device__ __forceinline__ void issue(const void* gsrc) {
        uint64_t ga = (uint64_t)(uintptr_t)gsrc;
        g0[2] = (uint32_t)ga;
        g0[3] = ((uint32_t)(ga >> 32) & 0x01FFFFFFu) | 0x80000000u; // addr hi | type=2
        asm volatile("tensor_load_to_lds %0, %1" :: "s"(g0), "s"(g1) : "memory");
    }
};

// ---------------------------------------------------------------------------
// One-time streaming fp32 -> bf16 convert (vectorized x4).
// ---------------------------------------------------------------------------
__global__ __launch_bounds__(256)
void convert_bf16(const float* __restrict__ src, unsigned short* __restrict__ dst, int n4)
{
    int i = blockIdx.x * 256 + threadIdx.x;
    if (i < n4) {
        float4 f = ((const float4*)src)[i];
        ushort4 o;
        o.x = f2bf(f.x); o.y = f2bf(f.y); o.z = f2bf(f.z); o.w = f2bf(f.w);
        ((ushort4*)dst)[i] = o;
    }
}

// ---------------------------------------------------------------------------
// One-time tiled transpose-convert: src[R][C] fp32 -> dst[C][R] bf16, per z.
// ---------------------------------------------------------------------------
__global__ __launch_bounds__(256)
void transpose_convert_bf16(const float* __restrict__ src, unsigned short* __restrict__ dst,
                            int R, int C)
{
    __shared__ float tile[32][33];
    const float* s = src + (size_t)blockIdx.z * R * C;
    unsigned short* d = dst + (size_t)blockIdx.z * R * C;
    int c0 = blockIdx.x * 32, r0 = blockIdx.y * 32;
    int tx = threadIdx.x & 31, ty = threadIdx.x >> 5;
#pragma unroll
    for (int i = 0; i < 4; ++i)
        tile[ty + i * 8][tx] = s[(size_t)(r0 + ty + i * 8) * C + c0 + tx];
    __syncthreads();
#pragma unroll
    for (int i = 0; i < 4; ++i)
        d[(size_t)(c0 + ty + i * 8) * R + r0 + tx] = f2bf(tile[tx][ty + i * 8]);
}

// ---------------------------------------------------------------------------
// C[M,N] = A[M,K] @ W[N,K]^T + bias[N].  bf16 operands, TDM-staged tiles,
// double-buffered LDS so the DMA overlaps the WMMAs.
// Block tile 128(M) x 128(N); 8 waves, each computes 32x64 (2x4 WMMA tiles).
// ---------------------------------------------------------------------------
template<bool OUT_BF16>
__global__ __launch_bounds__(256)
void gemm_bf16_wmma(const unsigned short* __restrict__ A, const unsigned short* __restrict__ W,
                    const float* __restrict__ bias, void* __restrict__ Cout,
                    int M, int N, int K)
{
    __shared__ __align__(16) unsigned short lds_a[2][128 * 32]; // [row][k]
    __shared__ __align__(16) unsigned short lds_b[2][128 * 32]; // [col][k]

    const int tid  = threadIdx.x;
    const int w    = tid >> 5;
    const int lane = tid & 31;
    const int half = lane >> 4;
    const int ln   = lane & 15;
    const int m0   = blockIdx.y * 128;
    const int n0   = blockIdx.x * 128;
    const int wm   = (w & 3) * 32;
    const int wn   = (w >> 2) * 64;

    const unsigned short* pa = A + (size_t)m0 * K;
    const unsigned short* pb = W + (size_t)n0 * K;

    Tdm2D da[2], db[2];
    da[0].init(lds_byte_addr(&lds_a[0][0]), 32, 128, (uint32_t)K);
    da[1].init(lds_byte_addr(&lds_a[1][0]), 32, 128, (uint32_t)K);
    db[0].init(lds_byte_addr(&lds_b[0][0]), 32, 128, (uint32_t)K);
    db[1].init(lds_byte_addr(&lds_b[1][0]), 32, 128, (uint32_t)K);

    v8f acc[2][4];
#pragma unroll
    for (int i = 0; i < 2; ++i)
#pragma unroll
        for (int j = 0; j < 4; ++j)
#pragma unroll
            for (int e = 0; e < 8; ++e) acc[i][j][e] = 0.0f;

    const int nk = K >> 5;
    if (w == 0)      da[0].issue(pa);
    else if (w == 1) db[0].issue(pb);
    __builtin_amdgcn_s_wait_tensorcnt(0);
    __syncthreads();

    for (int kc = 0; kc < nk; ++kc) {
        const int cur = kc & 1, nxt = cur ^ 1;
        if (kc + 1 < nk) {                 // prefetch next tile while computing
            if (w == 0)      da[nxt].issue(pa + (size_t)(kc + 1) * 32);
            else if (w == 1) db[nxt].issue(pb + (size_t)(kc + 1) * 32);
        }

        Frag16 af[2], bfg[4];
#pragma unroll
        for (int mi = 0; mi < 2; ++mi) {
            // A layout: lane row M=ln; e -> K = (e&7) + 8*half + 16*(e>>3)
            const unsigned short* p = &lds_a[cur][(wm + mi * 16 + ln) * 32 + 8 * half];
            af[mi].q[0] = *(const uint4*)p;
            af[mi].q[1] = *(const uint4*)(p + 16);
        }
#pragma unroll
        for (int ni = 0; ni < 4; ++ni) {
            // B layout: lane col N=ln; e -> K = (e&7) + 8*half + 16*(e>>3)
            const unsigned short* p = &lds_b[cur][(wn + ni * 16 + ln) * 32 + 8 * half];
            bfg[ni].q[0] = *(const uint4*)p;
            bfg[ni].q[1] = *(const uint4*)(p + 16);
        }
#pragma unroll
        for (int mi = 0; mi < 2; ++mi)
#pragma unroll
            for (int ni = 0; ni < 4; ++ni)
                acc[mi][ni] = __builtin_amdgcn_wmma_f32_16x16x32_bf16(
                    false, af[mi].v, false, bfg[ni].v, (short)0, acc[mi][ni], false, false);

        __builtin_amdgcn_s_wait_tensorcnt(0);  // publisher's DMA done before barrier
        __syncthreads();
    }

    // C/D layout: lane -> N = ln, reg v -> M = v + 8*half
#pragma unroll
    for (int mi = 0; mi < 2; ++mi)
#pragma unroll
        for (int ni = 0; ni < 4; ++ni)
#pragma unroll
            for (int v = 0; v < 8; ++v) {
                int rg = m0 + wm + mi * 16 + v + 8 * half;
                int cg = n0 + wn + ni * 16 + ln;
                float val = acc[mi][ni][v] + bias[cg];
                if (OUT_BF16)
                    ((unsigned short*)Cout)[(size_t)rg * N + cg] = f2bf(val);
                else
                    ((float*)Cout)[(size_t)rg * N + cg] = val;
            }
}

// ---------------------------------------------------------------------------
// Flash attention, analytic ALiBi + causal mask, bf16 operand flow,
// double-buffered TDM K/V staging.
// Qb: [B,S,E] bf16   Kb: [B,H,S,D] bf16   Vb: [B,H,D,S] bf16   Ob: [B,S,E] bf16
// Grid (S/128, H, B); 8 waves, each owns 16 query rows.
// ---------------------------------------------------------------------------
__global__ __launch_bounds__(256)
void attn_alibi_flash(const unsigned short* __restrict__ Qb, const unsigned short* __restrict__ Kb,
                      const unsigned short* __restrict__ Vb, unsigned short* __restrict__ Ob)
{
    __shared__ __align__(16) unsigned short lds_k[2][32 * 128];   // [key][d]
    __shared__ __align__(16) unsigned short lds_v[2][128 * 32];   // [d][key]
    __shared__ __align__(16) unsigned short lds_p[8][16 * 32];    // per-wave P staging

    const int tid  = threadIdx.x;
    const int w    = tid >> 5;
    const int lane = tid & 31;
    const int half = lane >> 4;
    const int ln   = lane & 15;
    const int qb0  = blockIdx.x * 128;
    const int h    = blockIdx.y;
    const int b    = blockIdx.z;
    const int qw   = qb0 + w * 16;

    const float scale = 0.08838834764831845f; // 1/sqrt(128)
    const float slope = exp2f(-0.5f * (float)(h + 1));

    // Q A-fragments (16 x 128): two aligned 16B loads per 32-wide chunk
    const unsigned short* Qrow = Qb + ((size_t)(b * SLEN + qw + ln)) * EMB + h * HD;
    Frag16 aq[4];
#pragma unroll
    for (int c = 0; c < 4; ++c) {
        const unsigned short* qp = Qrow + c * 32 + 8 * half;
        aq[c].q[0] = *(const uint4*)qp;
        aq[c].q[1] = *(const uint4*)(qp + 16);
    }

    const unsigned short* Kbase = Kb + ((size_t)(b * NH + h)) * SLEN * HD; // [s][d]
    const unsigned short* Vbase = Vb + ((size_t)(b * NH + h)) * HD * SLEN; // [d][s]

    Tdm2D dk[2], dv[2];
    dk[0].init(lds_byte_addr(&lds_k[0][0]), 4096, 1, 4096);   // 32x128 contiguous block
    dk[1].init(lds_byte_addr(&lds_k[1][0]), 4096, 1, 4096);
    dv[0].init(lds_byte_addr(&lds_v[0][0]), 32, 128, SLEN);   // 128 rows of 32 keys
    dv[1].init(lds_byte_addr(&lds_v[1][0]), 32, 128, SLEN);

    v8f o[8];
    float m[8], l[8], c0[8], c1[8];
    int qg_i[8];
#pragma unroll
    for (int dt = 0; dt < 8; ++dt)
#pragma unroll
        for (int e = 0; e < 8; ++e) o[dt][e] = 0.0f;
#pragma unroll
    for (int v = 0; v < 8; ++v) {
        int qg  = qw + v + 8 * half;
        qg_i[v] = qg;
        m[v]    = -3.0e38f;
        l[v]    = 0.0f;
        c0[v]   = slope * (float)(ln - qg);        // loop-invariant alibi parts
        c1[v]   = slope * (float)(ln + 16 - qg);
    }

    const int nkb = (qb0 >> 5) + 4;   // causal: keys 0 .. qb0+127
    if (w == 0)      dk[0].issue(Kbase);
    else if (w == 1) dv[0].issue(Vbase);
    __builtin_amdgcn_s_wait_tensorcnt(0);
    __syncthreads();

    for (int kb = 0; kb < nkb; ++kb) {
        const int cur = kb & 1, nxt = cur ^ 1;
        if (kb + 1 < nkb) {               // prefetch next K/V tiles during compute
            if (w == 0)      dk[nxt].issue(Kbase + (size_t)(kb + 1) * 32 * HD);
            else if (w == 1) dv[nxt].issue(Vbase + (size_t)(kb + 1) * 32);
        }

        if (kb * 32 <= qw + 15) {
            // ---- S = Q @ K^T : two 16x16 score tiles over 32 keys ----
            v8f s0, s1;
#pragma unroll
            for (int e = 0; e < 8; ++e) { s0[e] = 0.0f; s1[e] = 0.0f; }
#pragma unroll
            for (int c = 0; c < 4; ++c) {
                Frag16 k0, k1;
                const unsigned short* p0 = &lds_k[cur][(ln) * 128 + c * 32 + 8 * half];
                k0.q[0] = *(const uint4*)p0;
                k0.q[1] = *(const uint4*)(p0 + 16);
                const unsigned short* p1 = &lds_k[cur][(16 + ln) * 128 + c * 32 + 8 * half];
                k1.q[0] = *(const uint4*)p1;
                k1.q[1] = *(const uint4*)(p1 + 16);
                s0 = __builtin_amdgcn_wmma_f32_16x16x32_bf16(false, aq[c].v, false, k0.v,
                                                             (short)0, s0, false, false);
                s1 = __builtin_amdgcn_wmma_f32_16x16x32_bf16(false, aq[c].v, false, k1.v,
                                                             (short)0, s1, false, false);
            }

            // ---- hoisted alibi + causal mask + online softmax ----
            const float sk = slope * (float)(kb * 32);
            const int   k0i = kb * 32 + ln, k1i = k0i + 16;
            float p0a[8], p1a[8], corr[8];
#pragma unroll
            for (int v = 0; v < 8; ++v) {
                float x0 = s0[v] * scale + (sk + c0[v]);
                float x1 = s1[v] * scale + (sk + c1[v]);
                if (k0i > qg_i[v]) x0 = -65504.0f;
                if (k1i > qg_i[v]) x1 = -65504.0f;
                float rm = fmaxf(x0, x1);
                rm = fmaxf(rm, __shfl_xor(rm, 1));
                rm = fmaxf(rm, __shfl_xor(rm, 2));
                rm = fmaxf(rm, __shfl_xor(rm, 4));
                rm = fmaxf(rm, __shfl_xor(rm, 8));
                float mn = fmaxf(m[v], rm);
                float cr = __expf(m[v] - mn);
                float e0 = __expf(x0 - mn);
                float e1 = __expf(x1 - mn);
                float rs = e0 + e1;
                rs += __shfl_xor(rs, 1);
                rs += __shfl_xor(rs, 2);
                rs += __shfl_xor(rs, 4);
                rs += __shfl_xor(rs, 8);
                l[v] = l[v] * cr + rs;
                m[v] = mn;
                corr[v] = cr;
                p0a[v] = e0;
                p1a[v] = e1;
            }
#pragma unroll
            for (int dt = 0; dt < 8; ++dt)
#pragma unroll
                for (int v = 0; v < 8; ++v) o[dt][v] *= corr[v];

            // ---- P: C-layout -> A-layout via per-wave LDS ----
            unsigned short* pw = &lds_p[w][0];
#pragma unroll
            for (int v = 0; v < 8; ++v) {
                int row = v + 8 * half;
                pw[row * 32 + ln]      = f2bf(p0a[v]);
                pw[row * 32 + 16 + ln] = f2bf(p1a[v]);
            }
            Frag16 ap;
            const unsigned short* pp = &pw[ln * 32 + 8 * half];
            ap.q[0] = *(const uint4*)pp;
            ap.q[1] = *(const uint4*)(pp + 16);

            // ---- O += P @ V : 8 tiles across D=128 ----
#pragma unroll
            for (int dt = 0; dt < 8; ++dt) {
                Frag16 vf;
                const unsigned short* pv = &lds_v[cur][(dt * 16 + ln) * 32 + 8 * half];
                vf.q[0] = *(const uint4*)pv;
                vf.q[1] = *(const uint4*)(pv + 16);
                o[dt] = __builtin_amdgcn_wmma_f32_16x16x32_bf16(false, ap.v, false, vf.v,
                                                                (short)0, o[dt], false, false);
            }
        }

        __builtin_amdgcn_s_wait_tensorcnt(0);  // publisher DMA done before barrier
        __syncthreads();
    }

    // normalize (one reciprocal per row) + write merged-head bf16 output
    float inv[8];
#pragma unroll
    for (int v = 0; v < 8; ++v) inv[v] = 1.0f / l[v];
#pragma unroll
    for (int dt = 0; dt < 8; ++dt)
#pragma unroll
        for (int v = 0; v < 8; ++v) {
            int qg = qg_i[v];
            Ob[((size_t)(b * SLEN + qg)) * EMB + h * HD + dt * 16 + ln] =
                f2bf(o[dt][v] * inv[v]);
        }
}

extern "C" void kernel_launch(void* const* d_in, const int* in_sizes, int n_in,
                              void* d_out, int out_size, void* d_ws, size_t ws_size,
                              hipStream_t stream) {
    (void)in_sizes; (void)n_in; (void)out_size; (void)ws_size;
    const float* hidden = (const float*)d_in[0];
    const float* kst    = (const float*)d_in[1];   // [B,H,D,S]
    const float* vst    = (const float*)d_in[2];   // [B,H,S,D]
    // d_in[3] attention_mask, d_in[4] alibi_bias: computed analytically
    const float* Wq = (const float*)d_in[5];
    const float* bq = (const float*)d_in[6];
    const float* Wp = (const float*)d_in[7];
    const float* bp = (const float*)d_in[8];
    float* out = (float*)d_out;

    const size_t nTok = (size_t)NB * SLEN * EMB;   // 8.4M elems
    const size_t nW   = (size_t)EMB * EMB;         // 4.2M elems
    unsigned short* hbf  = (unsigned short*)d_ws;
    unsigned short* wqb  = hbf  + nTok;
    unsigned short* wpb  = wqb  + nW;
    unsigned short* kbf  = wpb  + nW;              // [B,H,S,D]
    unsigned short* vbf  = kbf  + nTok;            // [B,H,D,S]
    unsigned short* qbuf = vbf  + nTok;            // [B,S,E]
    unsigned short* obuf = qbuf + nTok;            // [B,S,E]

    // --- one-time bf16 pre-pass (streaming / tiled transpose) ---
    convert_bf16<<<(int)(nTok / 4 / 256), 256, 0, stream>>>(hidden, hbf, (int)(nTok / 4));
    convert_bf16<<<(int)(nW   / 4 / 256), 256, 0, stream>>>(Wq, wqb, (int)(nW / 4));
    convert_bf16<<<(int)(nW   / 4 / 256), 256, 0, stream>>>(Wp, wpb, (int)(nW / 4));
    transpose_convert_bf16<<<dim3(SLEN / 32, HD / 32, NB * NH), 256, 0, stream>>>(kst, kbf, HD, SLEN);
    transpose_convert_bf16<<<dim3(HD / 32, SLEN / 32, NB * NH), 256, 0, stream>>>(vst, vbf, SLEN, HD);

    // --- fused pipeline: q-proj -> flash attention -> out-proj ---
    dim3 gg(EMB / 128, (NB * SLEN) / 128);
    gemm_bf16_wmma<true><<<gg, 256, 0, stream>>>(hbf, wqb, bq, qbuf, NB * SLEN, EMB, EMB);

    dim3 ga(SLEN / 128, NH, NB);
    attn_alibi_flash<<<ga, 256, 0, stream>>>(qbuf, kbf, vbf, obuf);

    gemm_bf16_wmma<false><<<gg, 256, 0, stream>>>(obuf, wpb, bp, out, NB * SLEN, EMB, EMB);
}
